// PrototypicalNetwork_91070486544747
// MI455X (gfx1250) — matrix operand order repaired and checked by reading, not compiled, not run on hardware
//
#include <hip/hip_runtime.h>
#include <hip/hip_bf16.h>
#include <stdint.h>

// ---------------------------------------------------------------------------
// PrototypicalNetwork on MI455X (gfx1250):
//   feats = x @ W^T      (8192x4096 * 4096x2048)  -> bf16 WMMA, f32 accum
//   dots  = feats @ P^T  (8192x2048 * 2048x512)   -> bf16 WMMA, f32 accum
//   out   = -dots / (||feats|| * ||P||)
// Data path: pre-convert to bf16 in ws; TDM (tensor_load_to_lds) double-
// buffered into LDS with 16B/row padding; ds_load_b128 fragment reads;
// v_wmma_f32_16x16x32_bf16 compute; s_wait_tensorcnt + barriers for sync.
// ---------------------------------------------------------------------------

typedef __bf16 bf16_t;
typedef bf16_t v16bf __attribute__((ext_vector_type(16)));
typedef float  v8f   __attribute__((ext_vector_type(8)));
typedef unsigned int uint32x4 __attribute__((ext_vector_type(4)));
typedef int int32x4 __attribute__((ext_vector_type(4)));
typedef int int32x8 __attribute__((ext_vector_type(8)));

#define B_ROWS   8192
#define D_IN     4096
#define D_FEAT   2048
#define N_WAY    512
#define COS_EPS  1e-8f

// ---- workspace layout (bytes) ----
#define WS_XB   0ull
#define WS_WB   (WS_XB + (size_t)B_ROWS * D_IN * 2)     // x bf16      67108864
#define WS_PB   (WS_WB + (size_t)D_FEAT * D_IN * 2)     // W bf16     +16777216
#define WS_FB   (WS_PB + (size_t)N_WAY * D_FEAT * 2)    // P bf16      +2097152
#define WS_FSQ  (WS_FB + (size_t)B_ROWS * D_FEAT * 2)   // feats bf16 +33554432
#define WS_PN   (WS_FSQ + (size_t)B_ROWS * 4)           // feat_sq f32
// total ~119.6 MB

// ---- LDS staging: 128 rows x 64 bf16 (128B) + 16B TDM pad -> 144B stride ----
#define ROWB        144
#define STAGE_BYTES (128 * ROWB)        // 18432 per tile
#define BUF_BYTES   (2 * STAGE_BYTES)   // A tile + B tile
#define SMEM_BYTES  (2 * BUF_BYTES)     // double buffered = 73728 B

union Frag { uint32x4 q[2]; v16bf v; };

// Issue one TDM 2D tile load: tile_dim = 64 x 128 elements of 2 bytes,
// LDS padded 16B every 128B stored. Descriptor per CDNA5 ISA.
// This toolchain exposes the 6-arg builtin:
//   (uint32x4 g0, int32x8 g1, int32x4, int32x4, int32x8, i32 cpol)
// Only groups 0/1 carry state for a 2D tile (tile_dim2 == 0 -> unused dims).
__device__ __forceinline__ void tdm_load_2d(unsigned lds_off, uint64_t gaddr,
                                            unsigned tensor_d0, unsigned tensor_d1)
{
    uint32x4 g0;
    g0[0] = 1u;                                      // count=1, no gather
    g0[1] = lds_off;                                 // lds_addr (bytes)
    g0[2] = (unsigned)(gaddr & 0xFFFFFFFFull);       // global_addr[31:0]
    g0[3] = (unsigned)((gaddr >> 32) & 0x1FFFFFFull) // global_addr[56:32]
            | 0x80000000u;                           // type=2 ("image")
    int32x8 g1;
    // data_size=2B (code 1), pad_enable, pad_interval=128B (code 4),
    // pad_amount=16B (code 3)
    g1[0] = (int)((1u << 16) | (1u << 20) | (4u << 22) | (3u << 25));
    g1[1] = (int)((tensor_d0 & 0xFFFFu) << 16);                       // dim0[15:0]
    g1[2] = (int)((tensor_d0 >> 16) | ((tensor_d1 & 0xFFFFu) << 16)); // dim0[31:16],dim1[15:0]
    g1[3] = (int)((tensor_d1 >> 16) | (64u << 16));                   // dim1[31:16], tile_dim0=64
    g1[4] = (int)128u;                                                // tile_dim1=128
    g1[5] = (int)tensor_d0;                                           // dim0 stride (elements)
    g1[6] = 0;
    g1[7] = 0;
    int32x4 z4; z4[0] = 0; z4[1] = 0; z4[2] = 0; z4[3] = 0;           // groups 2/3 unused (2D)
    int32x8 z8;
    z8[0] = 0; z8[1] = 0; z8[2] = 0; z8[3] = 0;
    z8[4] = 0; z8[5] = 0; z8[6] = 0; z8[7] = 0;
    __builtin_amdgcn_tensor_load_to_lds(g0, g1, z4, z4, z8, 0);
}

// ---------------------------------------------------------------------------
// Kernel 1: f32 -> bf16 conversion of x, W, P + zero feat_sq accumulator.
// ---------------------------------------------------------------------------
__global__ __launch_bounds__(256)
void convert_zero_kernel(const float* __restrict__ x, const float* __restrict__ W,
                         const float* __restrict__ P,
                         bf16_t* __restrict__ xb, bf16_t* __restrict__ Wb,
                         bf16_t* __restrict__ Pb, float* __restrict__ feat_sq)
{
    const size_t NX = (size_t)B_ROWS * D_IN;
    const size_t NW = (size_t)D_FEAT * D_IN;
    const size_t NP = (size_t)N_WAY * D_FEAT;
    size_t i = (size_t)blockIdx.x * 256 + threadIdx.x;
    if (i < NX)                xb[i]           = (bf16_t)x[i];
    else if (i < NX + NW)      Wb[i - NX]      = (bf16_t)W[i - NX];
    else if (i < NX + NW + NP) Pb[i - NX - NW] = (bf16_t)P[i - NX - NW];
    if (i < B_ROWS) feat_sq[i] = 0.0f;
}

// ---------------------------------------------------------------------------
// Kernel 2: prototype norms from exact f32 data. One wave per row.
// ---------------------------------------------------------------------------
__global__ __launch_bounds__(256)
void proto_norm_kernel(const float* __restrict__ P, float* __restrict__ pn)
{
    const int wave = threadIdx.x >> 5;
    const int lane = threadIdx.x & 31;
    const int row  = blockIdx.x * 8 + wave;
    const float* p = P + (size_t)row * D_FEAT;
    float s = 0.0f;
    for (int j = lane; j < D_FEAT; j += 32) { float v = p[j]; s += v * v; }
    s += __shfl_xor(s, 1);
    s += __shfl_xor(s, 2);
    s += __shfl_xor(s, 4);
    s += __shfl_xor(s, 8);
    s += __shfl_xor(s, 16);
    if (lane == 0) pn[row] = fmaxf(sqrtf(s), COS_EPS);
}

// ---------------------------------------------------------------------------
// Kernel 3/4: tiled GEMM, C = A(MxK) * B(NxK)^T.  128x128 block tile,
// 8 waves (2x4), each wave 64x32 = 4x2 WMMA accumulators. TDM double buffer.
//   FINAL=false: store bf16 feats + atomically accumulate row sum-of-squares.
//   FINAL=true : out = -acc / (feat_norm * proto_norm), f32.
// ---------------------------------------------------------------------------
template <int KTOT, int NCOLS, bool FINAL>
__global__ __launch_bounds__(256, 1)
void gemm_tdm(const bf16_t* __restrict__ A, const bf16_t* __restrict__ Bm,
              bf16_t* __restrict__ Cbf, float* __restrict__ Cf,
              float* __restrict__ feat_sq, const float* __restrict__ proto_norm)
{
    extern __shared__ char smem[];
    const int tid    = threadIdx.x;
    const int wave   = tid >> 5;
    const int lane   = tid & 31;
    const int waveM  = wave >> 2;          // 0..1
    const int waveN  = wave & 3;           // 0..3
    const int blockN = blockIdx.x;
    const int blockM = blockIdx.y;
    const int m15    = lane & 15;
    const int hi     = lane >> 4;          // lane half

    v8f acc[4][2];
#pragma unroll
    for (int mt = 0; mt < 4; ++mt)
#pragma unroll
        for (int nt = 0; nt < 2; ++nt)
#pragma unroll
            for (int v = 0; v < 8; ++v) acc[mt][nt][v] = 0.0f;

    const int NSTAGE = KTOT / 64;
    const uint64_t gaA = (uint64_t)(uintptr_t)A  + (uint64_t)blockM * 128 * KTOT * 2;
    const uint64_t gaB = (uint64_t)(uintptr_t)Bm + (uint64_t)blockN * 128 * KTOT * 2;

    auto issue = [&](int kt) {
        const unsigned base = (unsigned)(kt & 1) * BUF_BYTES;
        const uint64_t koff = (uint64_t)kt * 64 * 2;
        tdm_load_2d(base,               gaA + koff, (unsigned)KTOT, (unsigned)B_ROWS);
        tdm_load_2d(base + STAGE_BYTES, gaB + koff, (unsigned)KTOT, (unsigned)NCOLS);
    };

    if (wave == 0) issue(0);

    for (int kt = 0; kt < NSTAGE; ++kt) {
        if (wave == 0) {
            if (kt + 1 < NSTAGE) {      // prefetch next stage, wait for current
                issue(kt + 1);
                __builtin_amdgcn_s_wait_tensorcnt((short)2);
            } else {
                __builtin_amdgcn_s_wait_tensorcnt((short)0);
            }
        }
        __syncthreads();                // stage kt resident in LDS for all waves

        const char* Ab = smem + (kt & 1) * BUF_BYTES;
        const char* Bb = Ab + STAGE_BYTES;

#pragma unroll
        for (int ks = 0; ks < 64; ks += 32) {
            Frag a[4], b[2];
#pragma unroll
            for (int mt = 0; mt < 4; ++mt) {
                // A frag 16x32 bf16: lane m=L%16, K halves offset 0/8 by lane half
                const char* rp = Ab + (waveM * 64 + mt * 16 + m15) * ROWB
                                    + (ks + hi * 8) * 2;
                a[mt].q[0] = *(const uint32x4*)rp;          // K = ks+ko .. +7
                a[mt].q[1] = *(const uint32x4*)(rp + 32);   // K = ks+16+ko .. +7
            }
#pragma unroll
            for (int nt = 0; nt < 2; ++nt) {
                // B frag 32x16 bf16: lane n=L%16, K halves offset 0/8 by lane half
                const char* rp = Bb + (waveN * 32 + nt * 16 + m15) * ROWB
                                    + (ks + hi * 8) * 2;
                b[nt].q[0] = *(const uint32x4*)rp;
                b[nt].q[1] = *(const uint32x4*)(rp + 32);
            }
#pragma unroll
            for (int mt = 0; mt < 4; ++mt)
#pragma unroll
                for (int nt = 0; nt < 2; ++nt)
                    acc[mt][nt] = __builtin_amdgcn_wmma_f32_16x16x32_bf16(
                        false, a[mt].v, false, b[nt].v,
                        (short)0, acc[mt][nt], false, false);
        }
        __syncthreads();                // all reads of buf done before reuse
    }

    // -------- epilogue --------
    const int rbase = blockM * 128 + waveM * 64 + hi * 8;
    const int cbase = blockN * 128 + waveN * 32 + m15;

    if (!FINAL) {
#pragma unroll
        for (int mt = 0; mt < 4; ++mt) {
#pragma unroll
            for (int v = 0; v < 8; ++v) {
                const int r = rbase + mt * 16 + v;
                const float s0 = acc[mt][0][v], s1 = acc[mt][1][v];
                Cbf[(size_t)r * D_FEAT + cbase]      = (bf16_t)s0;
                Cbf[(size_t)r * D_FEAT + cbase + 16] = (bf16_t)s1;
                float s = s0 * s0 + s1 * s1;        // reduce across the 16 cols
                s += __shfl_xor(s, 1);
                s += __shfl_xor(s, 2);
                s += __shfl_xor(s, 4);
                s += __shfl_xor(s, 8);
                if (m15 == 0) atomicAdd(&feat_sq[r], s);
            }
        }
    } else {
        const float ipn0 = 1.0f / proto_norm[cbase];
        const float ipn1 = 1.0f / proto_norm[cbase + 16];
#pragma unroll
        for (int mt = 0; mt < 4; ++mt) {
#pragma unroll
            for (int v = 0; v < 8; ++v) {
                const int r  = rbase + mt * 16 + v;
                const float fn  = fmaxf(sqrtf(feat_sq[r]), COS_EPS);
                const float inv = -1.0f / fn;
                Cf[(size_t)r * NCOLS + cbase]      = acc[mt][0][v] * inv * ipn0;
                Cf[(size_t)r * NCOLS + cbase + 16] = acc[mt][1][v] * inv * ipn1;
            }
        }
    }
}

// ---------------------------------------------------------------------------
extern "C" void kernel_launch(void* const* d_in, const int* in_sizes, int n_in,
                              void* d_out, int out_size, void* d_ws, size_t ws_size,
                              hipStream_t stream)
{
    (void)in_sizes; (void)n_in; (void)out_size; (void)ws_size;
    const float* x = (const float*)d_in[0];
    const float* W = (const float*)d_in[1];
    const float* P = (const float*)d_in[2];
    float* out = (float*)d_out;

    char* ws = (char*)d_ws;
    bf16_t* xb      = (bf16_t*)(ws + WS_XB);
    bf16_t* Wb      = (bf16_t*)(ws + WS_WB);
    bf16_t* Pb      = (bf16_t*)(ws + WS_PB);
    bf16_t* Fb      = (bf16_t*)(ws + WS_FB);
    float*  feat_sq = (float*)(ws + WS_FSQ);
    float*  proto_n = (float*)(ws + WS_PN);

    const size_t NTOT = (size_t)B_ROWS * D_IN + (size_t)D_FEAT * D_IN
                      + (size_t)N_WAY * D_FEAT;
    const int cvt_blocks = (int)((NTOT + 255) / 256);
    convert_zero_kernel<<<cvt_blocks, 256, 0, stream>>>(x, W, P, xb, Wb, Pb, feat_sq);
    proto_norm_kernel<<<N_WAY / 8, 256, 0, stream>>>(P, proto_n);

    // GEMM1: feats(bf16) = x @ W^T, plus row sum-of-squares
    gemm_tdm<D_IN, D_FEAT, false>
        <<<dim3(D_FEAT / 128, B_ROWS / 128), 256, SMEM_BYTES, stream>>>(
            xb, Wb, Fb, nullptr, feat_sq, nullptr);

    // GEMM2: out = -(feats @ P^T) / (||f|| * ||p||)
    gemm_tdm<D_FEAT, N_WAY, true>
        <<<dim3(N_WAY / 128, B_ROWS / 128), 256, SMEM_BYTES, stream>>>(
            Fb, Pb, nullptr, out, feat_sq, proto_n);
}
